// CrossAttnActGPT2Attention_2345052144050
// MI455X (gfx1250) — compile-verified, matrix operand order checked by reading, not run on hardware
//
#include <hip/hip_runtime.h>
#include <hip/hip_bf16.h>

// Problem constants (match reference)
#define B_  8
#define S_  4096
#define D_  1024
#define H_  16
#define HD_ 64

typedef __attribute__((ext_vector_type(16))) __bf16 v16bf;
typedef __attribute__((ext_vector_type(8)))  float  v8f;

union AB16 {
  v16bf v;
  __bf16 h[16];
  uint4 q[2];
};

__device__ __forceinline__ void pack8(__bf16* dst, float4 a, float4 b) {
  dst[0] = (__bf16)a.x; dst[1] = (__bf16)a.y; dst[2] = (__bf16)a.z; dst[3] = (__bf16)a.w;
  dst[4] = (__bf16)b.x; dst[5] = (__bf16)b.y; dst[6] = (__bf16)b.z; dst[7] = (__bf16)b.w;
}

// -------- Stage 0: kv = activation @ W_kv + b_kv  (8 x 2048, trivial) --------
__global__ void kv_gemv(const float* __restrict__ act,
                        const float* __restrict__ Wkv,
                        const float* __restrict__ bkv,
                        float* __restrict__ kvbuf) {
  const int b = blockIdx.x;
  const float* a = act + b * 1024;
  for (int c = threadIdx.x; c < 2048; c += blockDim.x) {
    float s = bkv[c];
    for (int i = 0; i < 1024; ++i) s += a[i] * Wkv[i * 2048 + c];
    kvbuf[b * 2048 + c] = s;
  }
}

// -------- Stage 1: effective keys / projected values ------------------------
// KeffT[b][col][d] = sum_t W_q[d][64h+t] * k_j[b,h,t]        (col = 2h+j)
// sbias[b][col]    = sum_t b_q[64h+t]   * k_j[b,h,t]
// PT[b][d][col]    = sum_t v_j[b,h,t]   * W_proj[64h+t][d]
__global__ void build_eff(const float* __restrict__ Wq,
                          const float* __restrict__ bq,
                          const float* __restrict__ nok,
                          const float* __restrict__ nov,
                          const float* __restrict__ Wproj,
                          const float* __restrict__ kvbuf,
                          float* __restrict__ sbias,
                          __bf16* __restrict__ keffT,
                          __bf16* __restrict__ ptT) {
  const int col = blockIdx.x;          // 0..31 == 2h + j
  const int b   = blockIdx.y;
  const int h = col >> 1, j = col & 1;
  __shared__ float kvec[64], vvec[64];
  if (threadIdx.x < 64) {
    int t = threadIdx.x;
    kvec[t] = j ? kvbuf[b * 2048 + h * 64 + t]        : nok[h * 64 + t];
    vvec[t] = j ? kvbuf[b * 2048 + 1024 + h * 64 + t] : nov[h * 64 + t];
  }
  __syncthreads();
  if (threadIdx.x == 0) {
    float s = 0.f;
    for (int t = 0; t < 64; ++t) s += bq[h * 64 + t] * kvec[t];
    sbias[b * 32 + col] = s;
  }
  for (int d = threadIdx.x; d < 1024; d += blockDim.x) {
    float ke = 0.f, pe = 0.f;
    for (int t = 0; t < 64; ++t) {
      ke += Wq[(size_t)d * 1024 + h * 64 + t] * kvec[t];
      pe += vvec[t] * Wproj[(size_t)(h * 64 + t) * 1024 + d];
    }
    keffT[((size_t)b * 32 + col) * 1024 + d] = (__bf16)ke;
    ptT[((size_t)b * 1024 + d) * 32 + col]   = (__bf16)pe;
  }
}

// -------- Stage 2: fused scores GEMM + softmax + output GEMM ----------------
// One wave (32 lanes) handles a 16-row tile of hidden_states end-to-end.
__global__ __launch_bounds__(128) void attn_main(
    const float* __restrict__ x,
    const float* __restrict__ bproj,
    const float* __restrict__ sbias,
    const __bf16* __restrict__ keffT,
    const __bf16* __restrict__ ptT,
    float* __restrict__ out) {
  __shared__ __align__(16) __bf16 ldsW[4][16][40];  // 40: pad, keeps rows 16B-aligned

  const int wv   = threadIdx.x >> 5;
  const int lane = threadIdx.x & 31;
  const int b    = blockIdx.y;
  const int tile = blockIdx.x * 4 + wv;      // 0..255 within batch
  const int s0   = tile * 16;
  const int row  = lane & 15;                // M-row for A loads / N-col for B & C
  const int hi   = lane >> 4;
  const int kb   = hi * 8;                   // K-group base per 16-bit WMMA striping

  const float*  xrow = x     + ((size_t)b * S_ + s0 + row) * D_;
  const __bf16* kr0  = keffT + ((size_t)b * 32 + row) * 1024;       // cols 0..15
  const __bf16* kr1  = keffT + ((size_t)b * 32 + 16 + row) * 1024;  // cols 16..31

  // ---- scores: (16 x 1024) @ (1024 x 32), two 16x16 C tiles -----------------
  v8f c0 = {}; v8f c1 = {};
  for (int ks = 0; ks < 32; ++ks) {
    const int kofs = ks * 32 + kb;
    if (ks < 31) __builtin_prefetch(xrow + kofs + 32, 0, 1);   // global_prefetch_b8

    AB16 a;
    float4 f0 = *(const float4*)(xrow + kofs);
    float4 f1 = *(const float4*)(xrow + kofs + 4);
    float4 f2 = *(const float4*)(xrow + kofs + 16);
    float4 f3 = *(const float4*)(xrow + kofs + 20);
    pack8(a.h,     f0, f1);      // K = kofs+0..7
    pack8(a.h + 8, f2, f3);      // K = kofs+16..23

    AB16 b0, b1;
    b0.q[0] = *(const uint4*)(kr0 + kofs);
    b0.q[1] = *(const uint4*)(kr0 + kofs + 16);
    b1.q[0] = *(const uint4*)(kr1 + kofs);
    b1.q[1] = *(const uint4*)(kr1 + kofs + 16);

    c0 = __builtin_amdgcn_wmma_f32_16x16x32_bf16(false, a.v, false, b0.v,
                                                 (short)0, c0, false, false);
    c1 = __builtin_amdgcn_wmma_f32_16x16x32_bf16(false, a.v, false, b1.v,
                                                 (short)0, c1, false, false);
  }

  // ---- softmax over adjacent column pairs (2h, 2h+1) ------------------------
  const float sb0 = sbias[b * 32 + row];
  const float sb1 = sbias[b * 32 + 16 + row];
#pragma unroll
  for (int r = 0; r < 8; ++r) {
    float s0v = c0[r] + sb0;
    float p0  = __shfl_xor(s0v, 1, 32);
    float m0  = fmaxf(s0v, p0);
    float e0  = __expf(s0v - m0), eo0 = __expf(p0 - m0);
    ldsW[wv][r + hi * 8][row]      = (__bf16)(e0 / (e0 + eo0));

    float s1v = c1[r] + sb1;
    float p1  = __shfl_xor(s1v, 1, 32);
    float m1  = fmaxf(s1v, p1);
    float e1  = __expf(s1v - m1), eo1 = __expf(p1 - m1);
    ldsW[wv][r + hi * 8][16 + row] = (__bf16)(e1 / (e1 + eo1));
  }
  __syncthreads();   // re-stripe C-layout weights -> A-layout via LDS

  // ---- output: (16 x 32) @ (32 x 1024) + b_proj -----------------------------
  AB16 aw;
  aw.q[0] = *(const uint4*)(&ldsW[wv][row][kb]);
  aw.q[1] = *(const uint4*)(&ldsW[wv][row][kb + 16]);

  for (int nt = 0; nt < 64; ++nt) {
    const int col = nt * 16 + row;
    const __bf16* pr = ptT + ((size_t)b * 1024 + col) * 32;
    AB16 bp;
    bp.q[0] = *(const uint4*)(pr + kb);
    bp.q[1] = *(const uint4*)(pr + kb + 16);

    v8f acc = {};
    acc = __builtin_amdgcn_wmma_f32_16x16x32_bf16(false, aw.v, false, bp.v,
                                                  (short)0, acc, false, false);
    const float bpv = bproj[col];
#pragma unroll
    for (int r = 0; r < 8; ++r) {
      out[((size_t)b * S_ + s0 + r + hi * 8) * D_ + col] = acc[r] + bpv;
    }
  }
}

extern "C" void kernel_launch(void* const* d_in, const int* in_sizes, int n_in,
                              void* d_out, int out_size, void* d_ws, size_t ws_size,
                              hipStream_t stream) {
  const float* hs  = (const float*)d_in[0];
  const float* act = (const float*)d_in[1];
  const float* Wq  = (const float*)d_in[2];
  const float* bq  = (const float*)d_in[3];
  const float* Wkv = (const float*)d_in[4];
  const float* bkv = (const float*)d_in[5];
  const float* nok = (const float*)d_in[6];
  const float* nov = (const float*)d_in[7];
  const float* Wp  = (const float*)d_in[8];
  const float* bp  = (const float*)d_in[9];
  float* out = (float*)d_out;

  // d_ws layout: kvbuf f32 [B*2D] | sbias f32 [B*32] | KeffT bf16 [B*32*1024] | PT bf16 [B*1024*32]
  float* kvbuf = (float*)d_ws;
  float* sbias = kvbuf + B_ * 2 * D_;
  __bf16* keffT = (__bf16*)(sbias + B_ * 32);
  __bf16* ptT   = keffT + (size_t)B_ * 32 * 1024;

  hipLaunchKernelGGL(kv_gemv,  dim3(B_),        dim3(256), 0, stream, act, Wkv, bkv, kvbuf);
  hipLaunchKernelGGL(build_eff, dim3(32, B_),   dim3(128), 0, stream,
                     Wq, bq, nok, nov, Wp, kvbuf, sbias, keffT, ptT);
  hipLaunchKernelGGL(attn_main, dim3(S_ / 64, B_), dim3(128), 0, stream,
                     hs, bp, sbias, keffT, ptT, out);
}
